// MultiViewSelfAttnProcessor_66864050864542
// MI455X (gfx1250) — matrix operand order repaired and checked by reading, not compiled, not run on hardware
//
#include <hip/hip_runtime.h>

typedef __bf16 bf16_t;
typedef __attribute__((ext_vector_type(16))) __bf16 v16bf;
typedef __attribute__((ext_vector_type(8)))  float  v8f;

struct __attribute__((aligned(16))) U128 { unsigned int w[4]; };
struct FragBits { U128 a, b; };

__device__ __forceinline__ U128 u128_zero() {
  U128 z; z.w[0] = z.w[1] = z.w[2] = z.w[3] = 0u; return z;
}
__device__ __forceinline__ v16bf frag_cast(U128 a, U128 b) {
  FragBits f{a, b};
  return __builtin_bit_cast(v16bf, f);
}

// A-operand fragment (16x32 bf16, row-major source): lane ln = row M.
// halves 0..7 -> k = hi*8 + j ; halves 8..15 -> k = 16 + hi*8 + j
__device__ __forceinline__ v16bf load_fragA(const bf16_t* rowp, int hi) {
  U128 a = *(const U128*)(rowp + hi * 8);
  U128 b = *(const U128*)(rowp + 16 + hi * 8);
  return frag_cast(a, b);
}
// Same, but only k = 0..15 valid (k = 16..31 zero-padded)
__device__ __forceinline__ v16bf load_fragA_tail16(const bf16_t* rowp, int hi) {
  U128 a = *(const U128*)(rowp + hi * 8);
  return frag_cast(a, u128_zero());
}
// B-operand fragment (32x16 bf16): lane ln = column N; colp = row-major row
// holding that column's data; halves 0..7 -> k = hi*8 + j ; halves 8..15 -> k = 16 + hi*8 + j
__device__ __forceinline__ v16bf load_fragB(const bf16_t* colp, int hi) {
  U128 a = *(const U128*)(colp + hi * 8);
  U128 b = *(const U128*)(colp + 16 + hi * 8);
  return frag_cast(a, b);
}
// Only k = 0..15 valid (k = 16..31 zero-padded)
__device__ __forceinline__ v16bf load_fragB_tail16(const bf16_t* colp, int hi) {
  U128 a = *(const U128*)(colp + hi * 8);
  return frag_cast(a, u128_zero());
}

__device__ __forceinline__ v8f wmma_bf16(v16bf a, v16bf b, v8f c) {
  return __builtin_amdgcn_wmma_f32_16x16x32_bf16(
      false, a, false, b, (short)0, c, false, false);
}

#define MDIM 4096
#define DDIM 640

// ---------------------------------------------------------------------------
// f32 -> bf16 cast, 4 elements per thread (b128 in, b64 out)
// ---------------------------------------------------------------------------
__global__ void cast_f32_bf16x4(const float* __restrict__ in, bf16_t* __restrict__ out, int n4) {
  int i = blockIdx.x * blockDim.x + threadIdx.x;
  if (i < n4) {
    float4 f = ((const float4*)in)[i];
    union { bf16_t h[4]; unsigned long long u; } u;
    u.h[0] = (bf16_t)f.x; u.h[1] = (bf16_t)f.y;
    u.h[2] = (bf16_t)f.z; u.h[3] = (bf16_t)f.w;
    ((unsigned long long*)out)[i] = u.u;
  }
}

// ---------------------------------------------------------------------------
// y = X @ W^T  (X: 4096x640 bf16, W: 640x640 bf16, both row-major).
// 128 threads = 4 waves; wave -> 64(M) x 16(N) strip.
// Fully unrolled, 2-stage software pipeline with two named register sets
// (no rotate copies -> no spills); compile-time dims -> base + IOFFSET addrs.
// outF != nullptr -> f32 output + bias; else bf16 output.
// ---------------------------------------------------------------------------
__global__ void gemm_xwT(const bf16_t* __restrict__ X, const bf16_t* __restrict__ W,
                         bf16_t* __restrict__ outB, float* __restrict__ outF,
                         const float* __restrict__ bias) {
  const int lane = threadIdx.x & 31;
  const int wave = threadIdx.x >> 5;
  const int ln = lane & 15;
  const int hi = lane >> 4;
  const int m0 = blockIdx.x * 64;
  const int n0 = blockIdx.y * 64 + wave * 16;

  v8f c[4];
  #pragma unroll
  for (int i = 0; i < 4; i++)
    #pragma unroll
    for (int r = 0; r < 8; r++) c[i][r] = 0.f;

  const bf16_t* wrow = W + (size_t)(n0 + ln) * DDIM;
  const bf16_t* xrow = X + (size_t)(m0 + ln) * DDIM;

  // stage 0 preload (k = 0)
  v16bf b0 = load_fragB(wrow, hi);
  v16bf a0_0 = load_fragA(xrow + 0 * 16 * DDIM, hi);
  v16bf a0_1 = load_fragA(xrow + 1 * 16 * DDIM, hi);
  v16bf a0_2 = load_fragA(xrow + 2 * 16 * DDIM, hi);
  v16bf a0_3 = load_fragA(xrow + 3 * 16 * DDIM, hi);

  #pragma unroll
  for (int k0 = 0; k0 < DDIM; k0 += 64) {
    // stage 1 preload (k = k0 + 32)
    v16bf b1 = load_fragB(wrow + k0 + 32, hi);
    v16bf a1_0 = load_fragA(xrow + 0 * 16 * DDIM + k0 + 32, hi);
    v16bf a1_1 = load_fragA(xrow + 1 * 16 * DDIM + k0 + 32, hi);
    v16bf a1_2 = load_fragA(xrow + 2 * 16 * DDIM + k0 + 32, hi);
    v16bf a1_3 = load_fragA(xrow + 3 * 16 * DDIM + k0 + 32, hi);

    c[0] = wmma_bf16(a0_0, b0, c[0]);
    c[1] = wmma_bf16(a0_1, b0, c[1]);
    c[2] = wmma_bf16(a0_2, b0, c[2]);
    c[3] = wmma_bf16(a0_3, b0, c[3]);

    if (k0 + 64 < DDIM) {  // stage 0 preload for next iteration (k = k0 + 64)
      b0   = load_fragB(wrow + k0 + 64, hi);
      a0_0 = load_fragA(xrow + 0 * 16 * DDIM + k0 + 64, hi);
      a0_1 = load_fragA(xrow + 1 * 16 * DDIM + k0 + 64, hi);
      a0_2 = load_fragA(xrow + 2 * 16 * DDIM + k0 + 64, hi);
      a0_3 = load_fragA(xrow + 3 * 16 * DDIM + k0 + 64, hi);
    }

    c[0] = wmma_bf16(a1_0, b1, c[0]);
    c[1] = wmma_bf16(a1_1, b1, c[1]);
    c[2] = wmma_bf16(a1_2, b1, c[2]);
    c[3] = wmma_bf16(a1_3, b1, c[3]);
  }

  #pragma unroll
  for (int i = 0; i < 4; i++) {
    size_t mbase = (size_t)(m0 + i * 16 + hi * 8);
    if (outF) {
      float bv = bias[n0 + ln];
      #pragma unroll
      for (int r = 0; r < 8; r++)
        outF[(mbase + r) * DDIM + n0 + ln] = c[i][r] + bv;
    } else {
      #pragma unroll
      for (int r = 0; r < 8; r++)
        outB[(mbase + r) * DDIM + n0 + ln] = (bf16_t)c[i][r];
    }
  }
}

// ---------------------------------------------------------------------------
// Multi-view flash attention, LDS-staged and double-buffered.
// Q/K/V: bf16 [4096][640]. Grid (16 m-blocks, 8 heads, 4 views), 128 threads.
// Per 32-key tile (shared by all 4 waves):
//   - K tile (32x80) staged row-major via global_load_async_to_lds_b128
//     (exactly 3 async instructions per wave -> ASYNCcnt-based double buffer)
//   - V tile staged transposed (V^T: 80x32) so P@V B-fragments are ds_load_b128
// ---------------------------------------------------------------------------
#define KSTR 88   // K LDS row stride (elements): 80 data + 8 pad, 176B rows
#define VSTR 40   // V^T LDS row stride (elements): 32 keys + 8 pad

__global__ void mv_flash_attn(const bf16_t* __restrict__ Q, const bf16_t* __restrict__ Kb,
                              const bf16_t* __restrict__ Vb, bf16_t* __restrict__ Ob) {
  __shared__ __attribute__((aligned(16))) bf16_t kbuf[2][32 * KSTR];
  __shared__ __attribute__((aligned(16))) bf16_t vtbuf[2][80 * VSTR];
  __shared__ __attribute__((aligned(16))) bf16_t pshare[4][16 * 16];

  const float scale = 0.11180339887498949f;  // 1/sqrt(80)
  const int tid = threadIdx.x;
  const int lane = tid & 31;
  const int wave = tid >> 5;
  const int ln = lane & 15;
  const int hi = lane >> 4;
  const int h0 = blockIdx.y * 80;
  const int q0 = blockIdx.z * 1024 + blockIdx.x * 64 + wave * 16;

  bf16_t* ps = &pshare[wave][0];

  // Q fragments (dh = 80 = 32 + 32 + 16tail), loaded once per wave.
  const bf16_t* qrow = Q + (size_t)(q0 + ln) * DDIM + h0;
  v16bf qf0 = load_fragA(qrow, hi);
  v16bf qf1 = load_fragA(qrow + 32, hi);
  v16bf qf2 = load_fragA_tail16(qrow + 64, hi);

  float mstate[8], lstate[8];
  #pragma unroll
  for (int r = 0; r < 8; r++) { mstate[r] = -1e30f; lstate[r] = 0.f; }
  v8f o[5];
  #pragma unroll
  for (int cix = 0; cix < 5; cix++)
    #pragma unroll
    for (int r = 0; r < 8; r++) o[cix][r] = 0.f;

  // Cooperative staging of one 32-key K/V tile into LDS buffer `buf`.
  auto stage = [&](int buf, int kr) {
    // --- K tile: 32 rows x 160B = 320 x 16B chunks, async copy, 3 instr/thread
    unsigned klds = (unsigned)(unsigned long long)(&kbuf[buf][0]);
    const bf16_t* kg = Kb + (size_t)kr * DDIM + h0;
    #pragma unroll
    for (int i = 0; i < 3; i++) {
      unsigned id = (i < 2) ? (unsigned)(i * 128 + tid) : (unsigned)(256 + (tid & 63));
      unsigned row = id / 10u;
      unsigned seg = id % 10u;
      unsigned laddr = klds + (row * (unsigned)KSTR + seg * 8u) * 2u;
      unsigned long long gaddr =
          (unsigned long long)(kg + (size_t)row * DDIM + (size_t)seg * 8);
      asm volatile("global_load_async_to_lds_b128 %0, %1, off"
                   :: "v"(laddr), "v"(gaddr) : "memory");
    }
    // --- V tile, transposed into V^T[80][32] (+pad): 20 elements per thread
    const bf16_t* vg = Vb + (size_t)kr * DDIM + h0;
    bf16_t* vt = &vtbuf[buf][0];
    #pragma unroll
    for (int i = 0; i < 20; i++) {
      unsigned e = (unsigned)(i * 128 + tid);   // 0..2559
      unsigned key = e / 80u;
      unsigned d = e % 80u;
      vt[d * VSTR + key] = vg[(size_t)key * DDIM + d];
    }
  };

  stage(0, 0);

  for (int t = 0; t < 128; t++) {
    const int buf = t & 1;
    if (t + 1 < 128) {
      stage(buf ^ 1, (t + 1) * 32);
      asm volatile("s_wait_asynccnt 3" ::: "memory");  // tile t's K copies done
    } else {
      asm volatile("s_wait_asynccnt 0" ::: "memory");
    }
    __syncthreads();  // staged tile visible to all waves

    const bf16_t* ktile = &kbuf[buf][0];
    const bf16_t* vt = &vtbuf[buf][0];

    #pragma unroll
    for (int st = 0; st < 2; st++) {          // two 16-key sub-tiles
      const bf16_t* krow = ktile + (size_t)(st * 16 + ln) * KSTR;
      v16bf kf0 = load_fragB(krow, hi);
      v16bf kf1 = load_fragB(krow + 32, hi);
      v16bf kf2 = load_fragB_tail16(krow + 64, hi);

      v8f s;
      #pragma unroll
      for (int r = 0; r < 8; r++) s[r] = 0.f;
      s = wmma_bf16(qf0, kf0, s);
      s = wmma_bf16(qf1, kf1, s);
      s = wmma_bf16(qf2, kf2, s);

      // Online softmax: row m = r + 8*hi lives across the 16 lanes of a group.
      float corr[8];
      #pragma unroll
      for (int r = 0; r < 8; r++) {
        float sv = s[r] * scale;
        float rm = sv;
        rm = fmaxf(rm, __shfl_xor(rm, 1, 32));
        rm = fmaxf(rm, __shfl_xor(rm, 2, 32));
        rm = fmaxf(rm, __shfl_xor(rm, 4, 32));
        rm = fmaxf(rm, __shfl_xor(rm, 8, 32));
        float mnew = fmaxf(mstate[r], rm);
        float cr = __expf(mstate[r] - mnew);
        float p = __expf(sv - mnew);
        float rs = p;
        rs += __shfl_xor(rs, 1, 32);
        rs += __shfl_xor(rs, 2, 32);
        rs += __shfl_xor(rs, 4, 32);
        rs += __shfl_xor(rs, 8, 32);
        lstate[r] = lstate[r] * cr + rs;
        mstate[r] = mnew;
        corr[r] = cr;
        s[r] = p;
      }
      #pragma unroll
      for (int cix = 0; cix < 5; cix++)
        #pragma unroll
        for (int r = 0; r < 8; r++) o[cix][r] *= corr[r];

      // P: C-layout -> A-layout via per-wave LDS bounce (keys padded 16->32).
      #pragma unroll
      for (int r = 0; r < 8; r++)
        ps[(r + 8 * hi) * 16 + ln] = (bf16_t)s[r];
      asm volatile("s_wait_dscnt 0" ::: "memory");
      v16bf pf = load_fragA_tail16(ps + ln * 16, hi);

      // O += P @ V : B columns of V chunk are V^T rows (contiguous in LDS).
      #pragma unroll
      for (int cix = 0; cix < 5; cix++) {
        const bf16_t* vrow = vt + (size_t)(cix * 16 + ln) * VSTR + st * 16;
        v16bf vf = load_fragB_tail16(vrow, hi);
        o[cix] = wmma_bf16(pf, vf, o[cix]);
      }
    }
    __syncthreads();  // all waves done with buf before it is restaged
  }

  // Normalize and store (bf16 feeds the O-projection GEMM).
  #pragma unroll
  for (int cix = 0; cix < 5; cix++) {
    #pragma unroll
    for (int r = 0; r < 8; r++) {
      float val = o[cix][r] / lstate[r];
      Ob[(size_t)(q0 + 8 * hi + r) * DDIM + h0 + cix * 16 + ln] = (bf16_t)val;
    }
  }
}

// ---------------------------------------------------------------------------
extern "C" void kernel_launch(void* const* d_in, const int* in_sizes, int n_in,
                              void* d_out, int out_size, void* d_ws, size_t ws_size,
                              hipStream_t stream) {
  (void)in_sizes; (void)n_in; (void)out_size; (void)ws_size;
  const float* x  = (const float*)d_in[0];
  const float* Wq = (const float*)d_in[1];
  const float* Wk = (const float*)d_in[2];
  const float* Wv = (const float*)d_in[3];
  const float* Wo = (const float*)d_in[4];
  const float* bo = (const float*)d_in[5];
  float* out = (float*)d_out;

  const size_t NT = (size_t)MDIM * DDIM;
  const size_t NW = (size_t)DDIM * DDIM;

  bf16_t* xb  = (bf16_t*)d_ws;
  bf16_t* wqb = xb + NT;
  bf16_t* wkb = wqb + NW;
  bf16_t* wvb = wkb + NW;
  bf16_t* wob = wvb + NW;
  bf16_t* qb  = wob + NW;
  bf16_t* kb  = qb + NT;
  bf16_t* vb  = kb + NT;
  bf16_t* ab  = vb + NT;

  const int nt4 = (int)(NT / 4), nw4 = (int)(NW / 4);
  cast_f32_bf16x4<<<dim3((nt4 + 255) / 256), 256, 0, stream>>>(x,  xb,  nt4);
  cast_f32_bf16x4<<<dim3((nw4 + 255) / 256), 256, 0, stream>>>(Wq, wqb, nw4);
  cast_f32_bf16x4<<<dim3((nw4 + 255) / 256), 256, 0, stream>>>(Wk, wkb, nw4);
  cast_f32_bf16x4<<<dim3((nw4 + 255) / 256), 256, 0, stream>>>(Wv, wvb, nw4);
  cast_f32_bf16x4<<<dim3((nw4 + 255) / 256), 256, 0, stream>>>(Wo, wob, nw4);

  dim3 gg(MDIM / 64, DDIM / 64);  // (64, 10)
  gemm_xwT<<<gg, 128, 0, stream>>>(xb, wqb, qb, nullptr, nullptr);
  gemm_xwT<<<gg, 128, 0, stream>>>(xb, wkb, kb, nullptr, nullptr);
  gemm_xwT<<<gg, 128, 0, stream>>>(xb, wvb, vb, nullptr, nullptr);

  mv_flash_attn<<<dim3(16, 8, 4), 128, 0, stream>>>(qb, kb, vb, ab);

  gemm_xwT<<<gg, 128, 0, stream>>>(ab, wob, nullptr, out, bo);
}
